// InnerSelfAttention_52475910423016
// MI455X (gfx1250) — matrix operand order repaired and checked by reading, not compiled, not run on hardware
//
#include <hip/hip_runtime.h>

// ---------- types ----------
typedef __attribute__((ext_vector_type(16))) __bf16        bf16x16;
typedef __attribute__((ext_vector_type(8)))  float         f32x8;
typedef __attribute__((ext_vector_type(4)))  float         f32x4;
typedef __attribute__((ext_vector_type(4)))  unsigned int  u32x4;
typedef __attribute__((ext_vector_type(2)))  unsigned int  u32x2;

union Frag { u32x4 q[2]; bf16x16 v; };

__device__ __forceinline__ unsigned short f2bf(float f) {
  unsigned u = __builtin_bit_cast(unsigned, f);
  u += 0x7fffu + ((u >> 16) & 1u);          // round-to-nearest-even
  return (unsigned short)(u >> 16);
}

// LDS offset of a __shared__ object: generic LDS pointer = SHARED_BASE | off,
// hardware maps flat->LDS via addr[31:0] (ISA 10.2), so truncation is exact.
__device__ __forceinline__ unsigned lds_addr(const void* p) {
  return (unsigned)(size_t)p;
}

// CDNA5 async copy: memory -> LDS, 16B per lane, tracked by ASYNCcnt.
__device__ __forceinline__ void async_copy_b128(unsigned lds_off, const void* g) {
  unsigned long long ga = (unsigned long long)(size_t)g;
  asm volatile("global_load_async_to_lds_b128 %0, %1, off"
               :: "v"(lds_off), "v"(ga) : "memory");
}
__device__ __forceinline__ void wait_async() {
  asm volatile("s_wait_asynccnt 0x0" ::: "memory");
}

// =====================================================================
// Pre-pass: f32 -> bf16 (8 elements / thread). n % 8 == 0.
// =====================================================================
__global__ __launch_bounds__(256) void f32_to_bf16_kernel(
    const float* __restrict__ in, unsigned short* __restrict__ out) {
  size_t i = ((size_t)blockIdx.x * 256 + threadIdx.x) * 8;
  f32x4 a = *(const f32x4*)(in + i);
  f32x4 b = *(const f32x4*)(in + i + 4);
  u32x4 p;
  p.x = (unsigned)f2bf(a.x) | ((unsigned)f2bf(a.y) << 16);
  p.y = (unsigned)f2bf(a.z) | ((unsigned)f2bf(a.w) << 16);
  p.z = (unsigned)f2bf(b.x) | ((unsigned)f2bf(b.y) << 16);
  p.w = (unsigned)f2bf(b.z) | ((unsigned)f2bf(b.w) << 16);
  *(u32x4*)(out + i) = p;
}

// =====================================================================
// GEMM: C[M,N] = A[M,K] * Bw[N,K]^T, A/Bw bf16, K contiguous for both.
// MODE 0: C = bf16, row-major                (Q, K projections)
// MODE 1: C = bf16, transposed per batch [b,n,s]   (V^T)
// MODE 2: C = f32 + bias, row-major          (output projection)
// 128x128 C tile / block; K-step 32; double-buffered async LDS tiles.
// =====================================================================
template <int MODE>
__global__ __launch_bounds__(256) void gemm_bf16_wmma(
    const unsigned short* __restrict__ Ab,
    const unsigned short* __restrict__ Bb,
    void* __restrict__ Cptr, const float* __restrict__ bias,
    int M, int N, int K, int S)
{
  __shared__ unsigned short As[2][128 * 40];   // 128 rows x 32 k (+8 pad)
  __shared__ unsigned short Bs[2][128 * 40];

  const int tid  = threadIdx.x;
  const int m0   = blockIdx.y * 128;
  const int n0   = blockIdx.x * 128;
  const int wid  = tid >> 5, lane = tid & 31;
  const int wm   = wid >> 1, wn = wid & 1;          // 4x2 wave grid
  const int l15  = lane & 15, half = lane >> 4;

  // per-thread async chunk coords: 512 x 16B chunks per tile, 2 per thread
  const int c0 = tid, c1 = tid + 256;
  const int r0 = c0 >> 2, kc0 = (c0 & 3) << 3;
  const int r1 = c1 >> 2, kc1 = (c1 & 3) << 3;

  auto issue_tile = [&](int buf, int k0) {
    async_copy_b128(lds_addr(&As[buf][r0 * 40 + kc0]),
                    Ab + (size_t)(m0 + r0) * K + k0 + kc0);
    async_copy_b128(lds_addr(&Bs[buf][r0 * 40 + kc0]),
                    Bb + (size_t)(n0 + r0) * K + k0 + kc0);
    async_copy_b128(lds_addr(&As[buf][r1 * 40 + kc1]),
                    Ab + (size_t)(m0 + r1) * K + k0 + kc1);
    async_copy_b128(lds_addr(&Bs[buf][r1 * 40 + kc1]),
                    Bb + (size_t)(n0 + r1) * K + k0 + kc1);
  };

  f32x8 acc[2][4];
#pragma unroll
  for (int mt = 0; mt < 2; ++mt)
#pragma unroll
    for (int nt = 0; nt < 4; ++nt) acc[mt][nt] = 0.0f;

  issue_tile(0, 0);
  wait_async();
  __syncthreads();

  int cur = 0;
  for (int k0 = 0; k0 < K; k0 += 32) {
    if (k0 + 32 < K) issue_tile(cur ^ 1, k0 + 32);   // overlaps with WMMAs

    Frag a[2], b[4];
#pragma unroll
    for (int mt = 0; mt < 2; ++mt) {
      const unsigned short* p =
          &As[cur][(wm * 32 + mt * 16 + l15) * 40 + half * 8];
      a[mt].q[0] = *(const u32x4*)p;
      a[mt].q[1] = *(const u32x4*)(p + 16);
    }
#pragma unroll
    for (int nt = 0; nt < 4; ++nt) {
      const unsigned short* p =
          &Bs[cur][(wn * 64 + nt * 16 + l15) * 40 + half * 8];
      b[nt].q[0] = *(const u32x4*)p;
      b[nt].q[1] = *(const u32x4*)(p + 16);
    }
#pragma unroll
    for (int mt = 0; mt < 2; ++mt)
#pragma unroll
      for (int nt = 0; nt < 4; ++nt)
        acc[mt][nt] = __builtin_amdgcn_wmma_f32_16x16x32_bf16(
            false, a[mt].v, false, b[nt].v, (short)0, acc[mt][nt],
            false, false);

    wait_async();        // next tile landed in LDS
    __syncthreads();     // all waves done reading cur
    cur ^= 1;
  }

  // ---- epilogue ----
#pragma unroll
  for (int mt = 0; mt < 2; ++mt)
#pragma unroll
    for (int nt = 0; nt < 4; ++nt)
#pragma unroll
      for (int v = 0; v < 8; ++v) {
        int row = m0 + wm * 32 + mt * 16 + half * 8 + v;   // C layout: VGPR=row
        int col = n0 + wn * 64 + nt * 16 + l15;            //           lane=col
        float val = acc[mt][nt][v];
        if (MODE == 0) {
          ((unsigned short*)Cptr)[(size_t)row * N + col] = f2bf(val);
        } else if (MODE == 1) {          // V^T: [b][n][s]
          int bb = row / S, s = row - bb * S;
          ((unsigned short*)Cptr)[(size_t)bb * N * S + (size_t)col * S + s] =
              f2bf(val);
        } else {
          ((float*)Cptr)[(size_t)row * N + col] = val + bias[col];
        }
      }
}

// =====================================================================
// Flash attention, causal. One block = (b,h, 64-query tile), 4 waves,
// each wave owns 16 query rows. K/V^T tiles async-staged in LDS, shared.
// =====================================================================
__global__ __launch_bounds__(128) void attn_flash_wmma(
    const unsigned short* __restrict__ Qb,
    const unsigned short* __restrict__ Kb,
    const unsigned short* __restrict__ Vt,
    unsigned short* __restrict__ Ob,
    int S, int E, int H)
{
  __shared__ unsigned short Qs[64 * 144];     // 64 q  x 128 d (+16 pad)
  __shared__ unsigned short Ks[64 * 144];     // 64 k  x 128 d
  __shared__ unsigned short Vs[128 * 72];     // 128 d x  64 s (+8 pad)
  __shared__ unsigned short Ps[4 * 16 * 72];  // per-wave 16 x 64 P tile

  const int tid = threadIdx.x;
  const int wid = tid >> 5, lane = tid & 31;
  const int l15 = lane & 15, half = lane >> 4;
  const int qt  = blockIdx.x;
  const int bh  = blockIdx.y;
  const int b   = bh / H, h = bh - b * H;

  const size_t qk_base = (size_t)b * S * E + (size_t)h * 128;

  // ---- stage Q tile once (async) ----
#pragma unroll
  for (int i = 0; i < 8; ++i) {
    int c = tid + i * 128;
    int r = c >> 4, v = c & 15;
    async_copy_b128(lds_addr(&Qs[r * 144 + v * 8]),
                    Qb + qk_base + (size_t)(qt * 64 + r) * E + v * 8);
  }

  float m_i[8], l_i[8];
  f32x8 o[8];
#pragma unroll
  for (int v = 0; v < 8; ++v) { m_i[v] = -3.0e38f; l_i[v] = 0.0f; }
#pragma unroll
  for (int t = 0; t < 8; ++t) o[t] = 0.0f;

  unsigned short* Pw = &Ps[wid * 16 * 72];
  const float sm   = 0.08838834764831845f;    // 1/sqrt(128)
  const int   qrow = qt * 64 + wid * 16 + half * 8;

  for (int kt = 0; kt <= qt; ++kt) {
    __syncthreads();
    // ---- async-stage K tile (64 x 128) and V^T tile (128 x 64) ----
    const size_t vt_base = (size_t)bh * 128 * S + (size_t)kt * 64;
#pragma unroll
    for (int i = 0; i < 8; ++i) {
      int c = tid + i * 128;
      int r = c >> 4, v = c & 15;
      async_copy_b128(lds_addr(&Ks[r * 144 + v * 8]),
                      Kb + qk_base + (size_t)(kt * 64 + r) * E + v * 8);
      int d = c >> 3, w = c & 7;
      async_copy_b128(lds_addr(&Vs[d * 72 + w * 8]),
                      Vt + vt_base + (size_t)d * S + w * 8);
    }
    wait_async();
    __syncthreads();

    // ---- S = Q K^T  (16x64, K-dim = 128 in 4 steps) ----
    f32x8 sc[4];
#pragma unroll
    for (int nt = 0; nt < 4; ++nt) sc[nt] = 0.0f;
#pragma unroll
    for (int ks = 0; ks < 4; ++ks) {
      Frag aq;
      const unsigned short* p = &Qs[(wid * 16 + l15) * 144 + ks * 32 + half * 8];
      aq.q[0] = *(const u32x4*)p;
      aq.q[1] = *(const u32x4*)(p + 16);
#pragma unroll
      for (int nt = 0; nt < 4; ++nt) {
        Frag bk;
        const unsigned short* pk = &Ks[(nt * 16 + l15) * 144 + ks * 32 + half * 8];
        bk.q[0] = *(const u32x4*)pk;
        bk.q[1] = *(const u32x4*)(pk + 16);
        sc[nt] = __builtin_amdgcn_wmma_f32_16x16x32_bf16(
            false, aq.v, false, bk.v, (short)0, sc[nt], false, false);
      }
    }

    // ---- scale, clamp to +/-1e4, causal mask ----
#pragma unroll
    for (int nt = 0; nt < 4; ++nt) {
      int kcol = kt * 64 + nt * 16 + l15;
#pragma unroll
      for (int v = 0; v < 8; ++v) {
        float s = sc[nt][v] * sm;
        s = fminf(fmaxf(s, -10000.0f), 10000.0f);
        if (kcol > qrow + v) s = -10000.0f;   // masked logits = -1e4 (ref clip)
        sc[nt][v] = s;
      }
    }

    // ---- online softmax (row = VGPR slot; reduce over 16-lane half) ----
    float tmax[8];
#pragma unroll
    for (int v = 0; v < 8; ++v)
      tmax[v] = fmaxf(fmaxf(sc[0][v], sc[1][v]), fmaxf(sc[2][v], sc[3][v]));
#pragma unroll
    for (int off = 1; off <= 8; off <<= 1)
#pragma unroll
      for (int v = 0; v < 8; ++v)
        tmax[v] = fmaxf(tmax[v], __shfl_xor(tmax[v], off, 32));

    float alpha[8], rsum[8];
#pragma unroll
    for (int v = 0; v < 8; ++v) {
      float mn = fmaxf(m_i[v], tmax[v]);
      alpha[v] = __expf(m_i[v] - mn);
      m_i[v]   = mn;
      rsum[v]  = 0.0f;
    }
#pragma unroll
    for (int nt = 0; nt < 4; ++nt)
#pragma unroll
      for (int v = 0; v < 8; ++v) {
        float p = __expf(sc[nt][v] - m_i[v]);
        rsum[v] += p;
        // C-layout -> A-layout via wave-private LDS (same-wave LDS in order)
        Pw[(half * 8 + v) * 72 + nt * 16 + l15] = f2bf(p);
      }
#pragma unroll
    for (int off = 1; off <= 8; off <<= 1)
#pragma unroll
      for (int v = 0; v < 8; ++v)
        rsum[v] += __shfl_xor(rsum[v], off, 32);
#pragma unroll
    for (int v = 0; v < 8; ++v) l_i[v] = l_i[v] * alpha[v] + rsum[v];
#pragma unroll
    for (int t = 0; t < 8; ++t)
#pragma unroll
      for (int v = 0; v < 8; ++v) o[t][v] *= alpha[v];

    // ---- O += P @ V  (K-dim = 64 keys in 2 steps, N = 128 d) ----
#pragma unroll
    for (int ks = 0; ks < 2; ++ks) {
      Frag ap;
      const unsigned short* pp = &Pw[l15 * 72 + ks * 32 + half * 8];
      ap.q[0] = *(const u32x4*)pp;
      ap.q[1] = *(const u32x4*)(pp + 16);
#pragma unroll
      for (int nt = 0; nt < 8; ++nt) {
        Frag bv;
        const unsigned short* pv = &Vs[(nt * 16 + l15) * 72 + ks * 32 + half * 8];
        bv.q[0] = *(const u32x4*)pv;
        bv.q[1] = *(const u32x4*)(pv + 16);
        o[nt] = __builtin_amdgcn_wmma_f32_16x16x32_bf16(
            false, ap.v, false, bv.v, (short)0, o[nt], false, false);
      }
    }
  }

  // ---- normalize and write merged-head output (bf16) ----
#pragma unroll
  for (int nt = 0; nt < 8; ++nt)
#pragma unroll
    for (int v = 0; v < 8; ++v) {
      float val = o[nt][v] / l_i[v];
      int q = qrow + v;
      int d = nt * 16 + l15;
      Ob[qk_base + (size_t)q * E + d] = f2bf(val);
    }
}

// =====================================================================
extern "C" void kernel_launch(void* const* d_in, const int* in_sizes, int n_in,
                              void* d_out, int out_size, void* d_ws, size_t ws_size,
                              hipStream_t stream) {
  const float* hs = (const float*)d_in[0];
  const float* Wq = (const float*)d_in[1];
  const float* Wk = (const float*)d_in[2];
  const float* Wv = (const float*)d_in[3];
  const float* Wo = (const float*)d_in[4];
  const float* bo = (const float*)d_in[5];
  float* out = (float*)d_out;

  const int B = 2, S = 2048, E = 2048, H = 16;
  const int M = B * S;
  const size_t ME = (size_t)M * E;   // 8.4M elems
  const size_t EE = (size_t)E * E;   // 4.2M elems

  unsigned short* Hb  = (unsigned short*)d_ws;   // bf16 hidden   [B,S,E]
  unsigned short* Wqb = Hb  + ME;                // bf16 weights  [E,E]
  unsigned short* Wkb = Wqb + EE;
  unsigned short* Wvb = Wkb + EE;
  unsigned short* Wob = Wvb + EE;
  unsigned short* Qb  = Wob + EE;                // bf16 [B,S,E]
  unsigned short* Kb  = Qb  + ME;                // bf16 [B,S,E]
  unsigned short* Vt  = Kb  + ME;                // bf16 [B,H,D,S]
  unsigned short* Ob  = Vt  + ME;                // bf16 [B,S,E]

  // ---- bf16 pre-pass (bandwidth-trivial at 23.3 TB/s) ----
  f32_to_bf16_kernel<<<dim3(ME / (8 * 256)), 256, 0, stream>>>(hs, Hb);
  f32_to_bf16_kernel<<<dim3(EE / (8 * 256)), 256, 0, stream>>>(Wq, Wqb);
  f32_to_bf16_kernel<<<dim3(EE / (8 * 256)), 256, 0, stream>>>(Wk, Wkb);
  f32_to_bf16_kernel<<<dim3(EE / (8 * 256)), 256, 0, stream>>>(Wv, Wvb);
  f32_to_bf16_kernel<<<dim3(EE / (8 * 256)), 256, 0, stream>>>(Wo, Wob);

  dim3 gg(E / 128, M / 128), gb(256);
  gemm_bf16_wmma<0><<<gg, gb, 0, stream>>>(Hb, Wqb, Qb, nullptr, M, E, E, S);
  gemm_bf16_wmma<0><<<gg, gb, 0, stream>>>(Hb, Wkb, Kb, nullptr, M, E, E, S);
  gemm_bf16_wmma<1><<<gg, gb, 0, stream>>>(Hb, Wvb, Vt, nullptr, M, E, E, S);

  attn_flash_wmma<<<dim3(S / 64, B * H), 128, 0, stream>>>(Qb, Kb, Vt, Ob, S, E, H);

  gemm_bf16_wmma<2><<<gg, gb, 0, stream>>>(Ob, Wob, out, bo, M, E, E, S);
}
